// TENChunkedLayer_7017976561857
// MI455X (gfx1250) — compile-verified, matrix operand order checked by reading, not run on hardware
//
#include <hip/hip_runtime.h>
#include <hip/hip_bf16.h>
#include <math.h>

// ---------------------------------------------------------------------------
// Problem constants: B=4, T=4096, D=1024, K=64, H=4, KH=16, C=64, MLP=4096.
// M = B*T = 16384 rows.
// ---------------------------------------------------------------------------
#define BB    4
#define TT    4096
#define DD    1024
#define KK    64
#define HH    4
#define KHH   16
#define CC    64
#define MLPD  4096
#define MROWS (BB * TT)

typedef __attribute__((ext_vector_type(16))) __bf16 v16bf;
typedef __attribute__((ext_vector_type(8)))  __bf16 v8bf;
typedef __attribute__((ext_vector_type(8)))  float  v8f;
typedef int v4i __attribute__((vector_size(16)));

#ifndef __has_builtin
#define __has_builtin(x) 0
#endif

#if __has_builtin(__builtin_amdgcn_global_load_async_to_lds_b128)
#define HAVE_ASYNC_LDS 1
#else
#define HAVE_ASYNC_LDS 0
#endif

typedef __attribute__((address_space(1))) v4i GV4;   // global int4
typedef __attribute__((address_space(3))) v4i LV4;   // LDS int4

// 16-byte global -> LDS copy.  Async (ASYNCcnt-tracked, no VGPR staging) when
// the gfx1250 builtin exists; otherwise a synchronous flat->ds fallback.
__device__ __forceinline__ void copy16_to_lds(const __bf16* g, __bf16* l) {
#if HAVE_ASYNC_LDS
    v4i* gv = (v4i*)const_cast<__bf16*>(g);   // pointee reinterpret (generic AS)
    v4i* lv = (v4i*)l;
    __builtin_amdgcn_global_load_async_to_lds_b128((GV4*)gv, (LV4*)lv, 0, 0);
#else
    *(v8bf*)l = *(const v8bf*)g;
#endif
}

template <int N>
__device__ __forceinline__ void wait_async() {
#if HAVE_ASYNC_LDS
#if __has_builtin(__builtin_amdgcn_s_wait_asynccnt)
    __builtin_amdgcn_s_wait_asynccnt(N);
#else
    asm volatile("s_wait_asynccnt %0" :: "n"(N) : "memory");
#endif
#endif
}

// ---------------------------------------------------------------------------
// fp32 -> bf16 elementwise convert (weights)
// ---------------------------------------------------------------------------
__global__ void cvt_bf16_kernel(const float* __restrict__ in,
                                __bf16* __restrict__ out, int n) {
    int i = blockIdx.x * 256 + threadIdx.x;
    if (i < n) out[i] = (__bf16)in[i];
}

// ---------------------------------------------------------------------------
// LayerNorm over D=1024, one block per row, 256 threads, 4 elems/thread.
// ---------------------------------------------------------------------------
__global__ __launch_bounds__(256) void ln_kernel(
    const float* __restrict__ x, const float* __restrict__ g,
    const float* __restrict__ b, __bf16* __restrict__ out) {
    const int row = blockIdx.x;
    const int tid = threadIdx.x;
    const float* xr = x + (size_t)row * DD;
    float v[4];
    float s = 0.f;
#pragma unroll
    for (int i = 0; i < 4; ++i) { v[i] = xr[tid + i * 256]; s += v[i]; }
    __shared__ float red[256];
    red[tid] = s; __syncthreads();
    for (int st = 128; st > 0; st >>= 1) {
        if (tid < st) red[tid] += red[tid + st];
        __syncthreads();
    }
    const float mean = red[0] * (1.f / DD);
    __syncthreads();
    float s2 = 0.f;
#pragma unroll
    for (int i = 0; i < 4; ++i) { float d = v[i] - mean; s2 += d * d; }
    red[tid] = s2; __syncthreads();
    for (int st = 128; st > 0; st >>= 1) {
        if (tid < st) red[tid] += red[tid + st];
        __syncthreads();
    }
    const float rstd = rsqrtf(red[0] * (1.f / DD) + 1e-5f);
#pragma unroll
    for (int i = 0; i < 4; ++i) {
        const int col = tid + i * 256;
        out[(size_t)row * DD + col] = (__bf16)((v[i] - mean) * rstd * g[col] + b[col]);
    }
}

// ---------------------------------------------------------------------------
// bf16 WMMA GEMM:  Out[M x N] = A[M x Kd] * W[N x Kd]^T
//
// Block tile 128x128, 256 threads = 8 waves (4 along M x 2 along N), wave
// tile 32x64 = 2x4 tiles of v_wmma_f32_16x16x32_bf16, fp32 accumulation.
//
// A and B 128x32 K-step tiles are staged in double-buffered LDS via
// global_load_async_to_lds_b128 (ASYNCcnt + s_wait_asynccnt + barrier),
// then fragments are read with ds_load_b128 in the ISA's striped bf16
// layout (lane half selects K offset +8; K halves at +0/+16).
// LDS rows padded to 40 bf16 (80 B) to spread banks.
//
// EPI: 0 = fp32 store; 1 = bias+SiLU -> bf16 store; 2 = bias+residual fp32.
// ---------------------------------------------------------------------------
template <int EPI>
__global__ __launch_bounds__(256) void gemm_bf16_wmma(
    const __bf16* __restrict__ A, const __bf16* __restrict__ W,
    float* __restrict__ outF, __bf16* __restrict__ outB,
    const float* __restrict__ bias, const float* __restrict__ resid,
    int Kd, int N) {
    constexpr int LROW = 40;                 // padded LDS row stride (elements)
    __shared__ __bf16 sA[2][128 * LROW];
    __shared__ __bf16 sB[2][128 * LROW];

    const int tid  = threadIdx.x;
    const int lane = tid & 31;
    const int wid  = tid >> 5;
    const int wm = wid & 3;                  // 4 waves along M
    const int wn = wid >> 2;                 // 2 waves along N
    const int m0 = blockIdx.y * 128;
    const int n0 = blockIdx.x * 128;
    const int half = lane >> 4;
    const int l15  = lane & 15;

    // Cooperative copy coords: 512 16B chunks per operand tile; thread t owns
    // chunks t and t+256 -> rows (t>>2) and (t>>2)+64, 16B quarter (t&3).
    const int crow = tid >> 2;
    const int cq   = tid & 3;
    const __bf16* gA = A + (size_t)(m0 + crow) * Kd + cq * 8;
    const __bf16* gW = W + (size_t)(n0 + crow) * Kd + cq * 8;
    const int lOff = crow * LROW + cq * 8;

    v8f acc[2][4];
    const v8f vzero = {0.f, 0.f, 0.f, 0.f, 0.f, 0.f, 0.f, 0.f};
#pragma unroll
    for (int i = 0; i < 2; ++i)
#pragma unroll
        for (int j = 0; j < 4; ++j) acc[i][j] = vzero;

    // prologue: stage first K-step into buffer 0
    {
        copy16_to_lds(gA, &sA[0][lOff]);
        copy16_to_lds(gA + (size_t)64 * Kd, &sA[0][lOff + 64 * LROW]);
        copy16_to_lds(gW, &sB[0][lOff]);
        copy16_to_lds(gW + (size_t)64 * Kd, &sB[0][lOff + 64 * LROW]);
    }

    int buf = 0;
    union Frag { v16bf v; v8bf h[2]; };
    for (int kk = 0; kk < Kd; kk += 32) {
        const bool more = (kk + 32) < Kd;
        if (more) {
            const int nb = buf ^ 1;
            const int kn = kk + 32;
            copy16_to_lds(gA + kn, &sA[nb][lOff]);
            copy16_to_lds(gA + (size_t)64 * Kd + kn, &sA[nb][lOff + 64 * LROW]);
            copy16_to_lds(gW + kn, &sB[nb][lOff]);
            copy16_to_lds(gW + (size_t)64 * Kd + kn, &sB[nb][lOff + 64 * LROW]);
            wait_async<4>();                 // previous buffer's copies done
        } else {
            wait_async<0>();
        }
        __syncthreads();

        Frag a[2], b[4];
        const __bf16* la = &sA[buf][(wm * 32 + l15) * LROW + half * 8];
        const __bf16* lb = &sB[buf][(wn * 64 + l15) * LROW + half * 8];
#pragma unroll
        for (int i = 0; i < 2; ++i) {
            const __bf16* p = la + i * 16 * LROW;
            a[i].h[0] = *(const v8bf*)p;
            a[i].h[1] = *(const v8bf*)(p + 16);
        }
#pragma unroll
        for (int j = 0; j < 4; ++j) {
            const __bf16* p = lb + j * 16 * LROW;
            b[j].h[0] = *(const v8bf*)p;
            b[j].h[1] = *(const v8bf*)(p + 16);
        }
#pragma unroll
        for (int i = 0; i < 2; ++i)
#pragma unroll
            for (int j = 0; j < 4; ++j)
                acc[i][j] = __builtin_amdgcn_wmma_f32_16x16x32_bf16(
                    false, a[i].v, false, b[j].v, (short)0, acc[i][j], false, false);
        __syncthreads();                     // tile consumed; safe to overwrite
        buf ^= 1;
    }

    // D layout: VGPR r -> M = r (lanes 0-15) / r+8 (lanes 16-31), N = lane&15.
#pragma unroll
    for (int i = 0; i < 2; ++i) {
#pragma unroll
        for (int j = 0; j < 4; ++j) {
            const int mbase = m0 + wm * 32 + i * 16 + half * 8;
            const int nb = n0 + wn * 64 + j * 16 + l15;
#pragma unroll
            for (int r = 0; r < 8; ++r) {
                const int m = mbase + r;
                const size_t idx = (size_t)m * N + nb;
                const float v = acc[i][j][r];
                if (EPI == 0) {
                    outF[idx] = v;
                } else if (EPI == 1) {
                    const float s = v + bias[nb];
                    outB[idx] = (__bf16)(s / (1.f + expf(-s)));   // SiLU
                } else {
                    outF[idx] = v + bias[nb] + resid[idx];
                }
            }
        }
    }
}

// ---------------------------------------------------------------------------
// Chunked complex eigenstate scan.  One 64-thread block per (k, b); thread j
// owns in-chunk position j.  o[j] = sum_{s<=j} decay^{j-s} * beta[s]
//                                    + decay^j * carry;  carry = o[63].
// beta/cbuf layout: (M x 128) fp32, cols [0,64)=real, [64,128)=imag.
// ---------------------------------------------------------------------------
__global__ __launch_bounds__(64) void eigen_scan_kernel(
    const float* __restrict__ beta, const float* __restrict__ log_decay,
    const float* __restrict__ freq, float* __restrict__ cbuf) {
    const int k = blockIdx.x;   // 0..63
    const int b = blockIdx.y;   // 0..3
    const int j = threadIdx.x;  // 0..63
    const float mag = 1.f / (1.f + expf(-log_decay[k]));
    const float f = freq[k];
    const float dc = mag * cosf(f);
    const float ds = mag * sinf(f);
    const float mj = powf(mag, (float)j);
    const float fcr = mj * cosf(f * (float)j);
    const float fci = mj * sinf(f * (float)j);

    __shared__ float sbr[CC], sbi[CC], scar[2];
    float carry_r = 0.f, carry_i = 0.f;

    for (int n = 0; n < TT / CC; ++n) {
        const size_t row = (size_t)(b * TT + n * CC + j) * 128;
        sbr[j] = beta[row + k];
        sbi[j] = beta[row + KK + k];
        __syncthreads();
        float ar = 0.f, ai = 0.f, wr = 1.f, wi = 0.f;
        for (int s = j; s >= 0; --s) {
            ar = fmaf(wr, sbr[s], fmaf(-wi, sbi[s], ar));
            ai = fmaf(wr, sbi[s], fmaf(wi, sbr[s], ai));
            const float nwr = wr * dc - wi * ds;
            wi = wr * ds + wi * dc;
            wr = nwr;
        }
        ar += fcr * carry_r - fci * carry_i;
        ai += fcr * carry_i + fci * carry_r;
        cbuf[row + k] = ar;
        cbuf[row + KK + k] = ai;
        __syncthreads();
        if (j == CC - 1) { scar[0] = ar; scar[1] = ai; }
        __syncthreads();
        carry_r = scar[0];
        carry_i = scar[1];
    }
}

// ---------------------------------------------------------------------------
// Per-head coupling: y[m, ri*64 + h*16 + j] = sum_k coup[h,j,k]*c[m, ri*64+h*16+k]
// One block per row (128 threads), output bf16 (A operand for w_out GEMM).
// ---------------------------------------------------------------------------
__global__ __launch_bounds__(128) void coupling_kernel(
    const float* __restrict__ c, const float* __restrict__ coup,
    __bf16* __restrict__ y) {
    const int row = blockIdx.x;
    const int col = threadIdx.x;         // 0..127
    __shared__ float sc[128];
    sc[col] = c[(size_t)row * 128 + col];
    __syncthreads();
    const int ri = col >> 6;
    const int loc = col & 63;
    const int h = loc >> 4;
    const int j = loc & 15;
    const float* cp = coup + (h * KHH + j) * KHH;
    const float* base = sc + ri * 64 + h * KHH;
    float s = 0.f;
#pragma unroll
    for (int k2 = 0; k2 < KHH; ++k2) s = fmaf(cp[k2], base[k2], s);
    y[(size_t)row * 128 + col] = (__bf16)s;
}

// ---------------------------------------------------------------------------
// Fused: x1 = x + sigmoid(gate_pre + b_gate)*h;  store x1 (fp32);
// LN2(x1) -> x2n (bf16).  One block per row.
// ---------------------------------------------------------------------------
__global__ __launch_bounds__(256) void gate_ln2_kernel(
    const float* __restrict__ x, const float* __restrict__ gate_pre,
    const float* __restrict__ h, const float* __restrict__ b_gate,
    const float* __restrict__ g2, const float* __restrict__ b2,
    float* __restrict__ x1, __bf16* __restrict__ x2n) {
    const int row = blockIdx.x;
    const int tid = threadIdx.x;
    const size_t base = (size_t)row * DD;
    float v[4];
    float s = 0.f;
#pragma unroll
    for (int i = 0; i < 4; ++i) {
        const int col = tid + i * 256;
        const float gp = gate_pre[base + col] + b_gate[col];
        const float gate = 1.f / (1.f + expf(-gp));
        const float x1v = x[base + col] + gate * h[base + col];
        x1[base + col] = x1v;
        v[i] = x1v;
        s += x1v;
    }
    __shared__ float red[256];
    red[tid] = s; __syncthreads();
    for (int st = 128; st > 0; st >>= 1) {
        if (tid < st) red[tid] += red[tid + st];
        __syncthreads();
    }
    const float mean = red[0] * (1.f / DD);
    __syncthreads();
    float s2 = 0.f;
#pragma unroll
    for (int i = 0; i < 4; ++i) { float d = v[i] - mean; s2 += d * d; }
    red[tid] = s2; __syncthreads();
    for (int st = 128; st > 0; st >>= 1) {
        if (tid < st) red[tid] += red[tid + st];
        __syncthreads();
    }
    const float rstd = rsqrtf(red[0] * (1.f / DD) + 1e-5f);
#pragma unroll
    for (int i = 0; i < 4; ++i) {
        const int col = tid + i * 256;
        x2n[base + col] = (__bf16)((v[i] - mean) * rstd * g2[col] + b2[col]);
    }
}

// ---------------------------------------------------------------------------
// Host-side orchestration
// ---------------------------------------------------------------------------
extern "C" void kernel_launch(void* const* d_in, const int* in_sizes, int n_in,
                              void* d_out, int out_size, void* d_ws, size_t ws_size,
                              hipStream_t stream) {
    (void)in_sizes; (void)n_in; (void)out_size; (void)ws_size;
    const float* x        = (const float*)d_in[0];
    const float* w_in     = (const float*)d_in[1];
    const float* w_out    = (const float*)d_in[2];
    const float* w_gate   = (const float*)d_in[3];
    const float* b_gate   = (const float*)d_in[4];
    const float* log_dec  = (const float*)d_in[5];
    const float* freq     = (const float*)d_in[6];
    const float* coup     = (const float*)d_in[7];
    const float* w_mlp1   = (const float*)d_in[8];
    const float* b_mlp1   = (const float*)d_in[9];
    const float* w_mlp2   = (const float*)d_in[10];
    const float* b_mlp2   = (const float*)d_in[11];
    const float* ln1_g    = (const float*)d_in[12];
    const float* ln1_b    = (const float*)d_in[13];
    const float* ln2_g    = (const float*)d_in[14];
    const float* ln2_b    = (const float*)d_in[15];
    float* outp = (float*)d_out;
    char*  ws   = (char*)d_ws;

    size_t off = 0;
    auto alloc = [&](size_t bytes) {
        size_t o = off;
        off += (bytes + 255) & ~(size_t)255;
        return o;
    };
    const size_t M = MROWS;
    const size_t off_xnorm  = alloc(M * DD * 2);            // bf16
    const size_t off_win    = alloc((size_t)2 * KK * DD * 2);
    const size_t off_wgate  = alloc((size_t)DD * DD * 2);
    const size_t off_wout   = alloc((size_t)DD * 2 * KK * 2);
    const size_t off_wmlp1  = alloc((size_t)MLPD * DD * 2);
    const size_t off_wmlp2  = alloc((size_t)DD * MLPD * 2);
    const size_t off_beta   = alloc(M * 128 * 4);           // fp32
    const size_t off_cbuf   = alloc(M * 128 * 4);           // fp32
    const size_t off_ybf    = alloc(M * 128 * 2);           // bf16
    const size_t off_gpre   = alloc(M * DD * 4);            // fp32 (64 MB)
    const size_t off_hbuf   = alloc(M * DD * 4);            // fp32 (adjacent 64 MB)
    const size_t off_x1     = alloc(M * DD * 4);            // fp32
    const size_t off_x2n    = alloc(M * DD * 2);            // bf16
    // mlp1 activation (bf16, 128 MB) reuses the dead gpre+hbuf region
    const size_t off_mlp1a  = off_gpre;

    __bf16* xnorm_bf = (__bf16*)(ws + off_xnorm);
    __bf16* win_bf   = (__bf16*)(ws + off_win);
    __bf16* wgate_bf = (__bf16*)(ws + off_wgate);
    __bf16* wout_bf  = (__bf16*)(ws + off_wout);
    __bf16* wmlp1_bf = (__bf16*)(ws + off_wmlp1);
    __bf16* wmlp2_bf = (__bf16*)(ws + off_wmlp2);
    float*  beta     = (float*)(ws + off_beta);
    float*  cbuf     = (float*)(ws + off_cbuf);
    __bf16* ybf      = (__bf16*)(ws + off_ybf);
    float*  gpre     = (float*)(ws + off_gpre);
    float*  hbuf     = (float*)(ws + off_hbuf);
    float*  x1       = (float*)(ws + off_x1);
    __bf16* x2n      = (__bf16*)(ws + off_x2n);
    __bf16* mlp1a    = (__bf16*)(ws + off_mlp1a);

    // 1) LN1 -> xnorm (bf16)
    ln_kernel<<<M, 256, 0, stream>>>(x, ln1_g, ln1_b, xnorm_bf);

    // 2) weight conversions fp32 -> bf16
    auto cvt = [&](const float* src, __bf16* dst, int n) {
        cvt_bf16_kernel<<<(n + 255) / 256, 256, 0, stream>>>(src, dst, n);
    };
    cvt(w_in,   win_bf,   2 * KK * DD);
    cvt(w_gate, wgate_bf, DD * DD);
    cvt(w_out,  wout_bf,  DD * 2 * KK);
    cvt(w_mlp1, wmlp1_bf, MLPD * DD);
    cvt(w_mlp2, wmlp2_bf, DD * MLPD);

    // 3) beta = xnorm @ w_in^T            (M x 128, Kd = 1024)
    gemm_bf16_wmma<0><<<dim3(128 / 128, M / 128), 256, 0, stream>>>(
        xnorm_bf, win_bf, beta, nullptr, nullptr, nullptr, DD, 128);

    // 4) chunked complex eigenstate scan  (cbuf = concat(c_r, c_i))
    eigen_scan_kernel<<<dim3(KK, BB), CC, 0, stream>>>(beta, log_dec, freq, cbuf);

    // 5) per-head coupling -> y (bf16)
    coupling_kernel<<<M, 128, 0, stream>>>(cbuf, coup, ybf);

    // 6) h = y @ w_out^T                  (M x 1024, Kd = 128)
    gemm_bf16_wmma<0><<<dim3(DD / 128, M / 128), 256, 0, stream>>>(
        ybf, wout_bf, hbuf, nullptr, nullptr, nullptr, 128, DD);

    // 7) gate_pre = xnorm @ w_gate^T      (M x 1024, Kd = 1024)
    gemm_bf16_wmma<0><<<dim3(DD / 128, M / 128), 256, 0, stream>>>(
        xnorm_bf, wgate_bf, gpre, nullptr, nullptr, nullptr, DD, DD);

    // 8) x1 = x + sigmoid(gate_pre+b)*h ; LN2 -> x2n (bf16)
    gate_ln2_kernel<<<M, 256, 0, stream>>>(x, gpre, hbuf, b_gate, ln2_g, ln2_b, x1, x2n);

    // 9) mlp1a = silu(x2n @ w_mlp1^T + b_mlp1)  (M x 4096, Kd = 1024, bf16 out)
    gemm_bf16_wmma<1><<<dim3(MLPD / 128, M / 128), 256, 0, stream>>>(
        x2n, wmlp1_bf, nullptr, mlp1a, b_mlp1, nullptr, DD, MLPD);

    // 10) out = x1 + mlp1a @ w_mlp2^T + b_mlp2  (M x 1024, Kd = 4096)
    gemm_bf16_wmma<2><<<dim3(DD / 128, M / 128), 256, 0, stream>>>(
        mlp1a, wmlp2_bf, outp, nullptr, b_mlp2, x1, MLPD, DD);
}
